// Attention_16982300688391
// MI455X (gfx1250) — matrix operand order repaired and checked by reading, not compiled, not run on hardware
//
#include <hip/hip_runtime.h>
#include <hip/hip_bf16.h>
#include <stdint.h>

// Problem constants (match reference)
#define B_   8
#define T_   1024
#define C_   1024
#define H_   16
#define HKV_ 4
#define HD_  64
#define NKV_ 512           // 2*C/G
#define CAP_ 50.0f
#define EPS_ 1e-6f

typedef __attribute__((ext_vector_type(16))) __bf16 v16bf;
typedef __attribute__((ext_vector_type(8)))  float  v8f;
typedef __attribute__((ext_vector_type(4)))  unsigned int v4u;
typedef __attribute__((ext_vector_type(8)))  int v8i;
typedef __attribute__((ext_vector_type(4)))  int v4i;

static __device__ __forceinline__ v8f wmma_bf16(v16bf a, v16bf b, v8f c) {
  // D = A(16x32 bf16) x B(32x16 bf16) + C(16x16 f32)
  return __builtin_amdgcn_wmma_f32_16x16x32_bf16(
      false, a, false, b, (short)0, c, false, false);
}

// ---- WMMA fragment loaders (wave32 layouts per CDNA5 ISA 7.12.2) ----
static __device__ __forceinline__ v16bf load_frag_a(const __bf16* p, int ld) {
  const int lane = threadIdx.x & 31;
  const int m = lane & 15, hi = lane >> 4;
  const uint32_t* s0 = (const uint32_t*)(p + (size_t)m * ld + hi * 8);
  const uint32_t* s1 = (const uint32_t*)(p + (size_t)m * ld + 16 + hi * 8);
  union { v16bf v; uint32_t u[8]; } f;
#pragma unroll
  for (int i = 0; i < 4; ++i) { f.u[i] = s0[i]; f.u[4 + i] = s1[i]; }
  return f.v;
}

static __device__ __forceinline__ v16bf load_frag_b(const __bf16* p, int ld) {
  const int lane = threadIdx.x & 31;
  const int n = lane & 15, hi = lane >> 4;
  const uint32_t* s0 = (const uint32_t*)(p + (size_t)n * ld + hi * 8);
  const uint32_t* s1 = (const uint32_t*)(p + (size_t)n * ld + 16 + hi * 8);
  union { v16bf v; uint32_t u[8]; } f;
#pragma unroll
  for (int i = 0; i < 4; ++i) { f.u[i] = s0[i]; f.u[4 + i] = s1[i]; }
  return f.v;
}

// ---- Tensor Data Mover: 2-D tile (tile_d0 x tile_d1 elems, 2-byte data)
// from row-major global tensor into LDS, per CDNA5 ISA ch.8 D# layout. ----
static __device__ __forceinline__ uint32_t lds_addr32(const void* p) {
  // LDS aperture base has zero low 32 bits -> low word of generic address
  // is the wave-relative LDS byte offset.
  return (uint32_t)(uintptr_t)p;
}

static __device__ __forceinline__ void tdm_load_2d(
    const __bf16* gtile, uint32_t lds_byte,
    uint32_t tensor_d0, uint32_t tensor_d1, uint32_t stride0_elems,
    uint32_t tile_d0, uint32_t tile_d1) {
  const uint64_t ga = (uint64_t)(uintptr_t)gtile;
  v4u g0;
  g0[0] = 1u;                                     // count=1, no gather
  g0[1] = lds_byte;                               // lds_addr
  g0[2] = (uint32_t)ga;                           // global_addr[31:0]
  g0[3] = (uint32_t)((ga >> 32) & 0x01FFFFFFu) |  // global_addr[56:32]
          (2u << 30);                              // type=2 ("image")
  v8i g1;
  g1[0] = (int)(1u << 16);                        // data_size=1 -> 2 bytes
  g1[1] = (int)((tensor_d0 & 0xFFFFu) << 16);     // tensor_dim0[15:0]
  g1[2] = (int)((tensor_d0 >> 16) | ((tensor_d1 & 0xFFFFu) << 16));
  g1[3] = (int)((tensor_d1 >> 16) | ((tile_d0 & 0xFFFFu) << 16));
  g1[4] = (int)(tile_d1 & 0xFFFFu);               // tile_dim1 (tile_dim2=0)
  g1[5] = (int)stride0_elems;                     // tensor_dim0_stride[31:0]
  g1[6] = 0;                                      // stride0 hi / stride1 lo
  g1[7] = 0;
  v4i z4; z4[0] = z4[1] = z4[2] = z4[3] = 0;      // 2-D: groups 2/3 unused
  v8i z8;
#pragma unroll
  for (int i = 0; i < 8; ++i) z8[i] = 0;
  __builtin_amdgcn_tensor_load_to_lds(g0, g1, z4, z4, z8, 0);
}

// ---------------- fp32 -> bf16 convert (4-wide) ----------------
__global__ void cvt_f32_bf16(const float* __restrict__ in,
                             __bf16* __restrict__ out, int n4) {
  int i = blockIdx.x * blockDim.x + threadIdx.x;
  if (i < n4) {
    float4 v = ((const float4*)in)[i];
    __bf16* o = out + (size_t)i * 4;
    o[0] = (__bf16)v.x; o[1] = (__bf16)v.y;
    o[2] = (__bf16)v.z; o[3] = (__bf16)v.w;
  }
}

// ------- Tiled WMMA GEMM, TDM-fed & double buffered: out = A[M,K]*W[N,K]^T
// 256 threads (8 waves); block tile 128x128; wave tile 32x64 (8 WMMA/k-step).
__global__ __launch_bounds__(256)
void gemm_bf16_wmma(const __bf16* __restrict__ A, const __bf16* __restrict__ W,
                    float* __restrict__ out, int M, int N, int K) {
  __shared__ __bf16 sA[2][128][32];
  __shared__ __bf16 sB[2][128][32];
  const int tid  = threadIdx.x;
  const int wave = tid >> 5;
  const int wm = wave & 3;      // 4 M sub-tiles of 32 rows
  const int wn = wave >> 2;     // 2 N sub-tiles of 64 cols
  const int n0 = blockIdx.x * 128;
  const int m0 = blockIdx.y * 128;

  const v8f vzero = {0.f,0.f,0.f,0.f,0.f,0.f,0.f,0.f};
  v8f acc[2][4];
#pragma unroll
  for (int i = 0; i < 2; ++i)
#pragma unroll
    for (int j = 0; j < 4; ++j) acc[i][j] = vzero;

  const int nk = K / 32;
  // Prologue: wave 0 launches TDM for stage 0 into buffer 0.
  if (tid < 32) {
    tdm_load_2d(A + (size_t)m0 * K, lds_addr32(&sA[0][0][0]), K, M, K, 32, 128);
    tdm_load_2d(W + (size_t)n0 * K, lds_addr32(&sB[0][0][0]), K, N, K, 32, 128);
  }
  for (int s = 0; s < nk; ++s) {
    const int cur = s & 1;
    if (tid < 32) {
      if (s + 1 < nk) {
        const int k1 = (s + 1) * 32;
        tdm_load_2d(A + (size_t)m0 * K + k1, lds_addr32(&sA[cur ^ 1][0][0]),
                    K, M, K, 32, 128);
        tdm_load_2d(W + (size_t)n0 * K + k1, lds_addr32(&sB[cur ^ 1][0][0]),
                    K, N, K, 32, 128);
        __builtin_amdgcn_s_wait_tensorcnt(2);   // stage s retired (in-order)
      } else {
        __builtin_amdgcn_s_wait_tensorcnt(0);
      }
    }
    __syncthreads();                            // LDS tiles visible to all
    const __bf16* aptr = &sA[cur][wm * 32][0];
    const __bf16* bptr = &sB[cur][wn * 64][0];
    v16bf af0 = load_frag_a(aptr, 32);
    v16bf af1 = load_frag_a(aptr + 16 * 32, 32);
#pragma unroll
    for (int j = 0; j < 4; ++j) {
      v16bf bfm = load_frag_b(bptr + j * 16 * 32, 32);
      acc[0][j] = wmma_bf16(af0, bfm, acc[0][j]);
      acc[1][j] = wmma_bf16(af1, bfm, acc[1][j]);
    }
    __syncthreads();                            // done reading before overwrite
  }

  const int lane = tid & 31;
  const int ln = lane & 15, hi = lane >> 4;
#pragma unroll
  for (int i = 0; i < 2; ++i)
#pragma unroll
    for (int j = 0; j < 4; ++j)
#pragma unroll
      for (int r = 0; r < 8; ++r)
        out[(size_t)(m0 + wm * 32 + i * 16 + hi * 8 + r) * N +
            (n0 + wn * 64 + j * 16 + ln)] = acc[i][j][r];
}

// -------- Fused RMSNorm + RoPE + layout transform (one wave per vector) ----
__global__ __launch_bounds__(256)
void norm_rope_pack(const float* __restrict__ qraw,
                    const float* __restrict__ kvraw,
                    const float* __restrict__ qw, const float* __restrict__ kw,
                    __bf16* __restrict__ Qo, __bf16* __restrict__ Ko,
                    __bf16* __restrict__ Vt) {
  const int wave = threadIdx.x >> 5, lane = threadIdx.x & 31;
  const long id = (long)blockIdx.x * 8 + wave;
  const long NQ = (long)B_ * T_ * H_;
  const long NK = (long)B_ * T_ * HKV_;
  if (id >= NQ + NK) return;

  float v1, v2, w1, w2;
  int t;
  __bf16* dst;
  int b;
  if (id < NQ) {
    int h = (int)(id % H_); long bt = id / H_;
    t = (int)(bt % T_); b = (int)(bt / T_);
    const float* src = qraw + ((size_t)b * T_ + t) * C_ + h * HD_;
    v1 = src[lane]; v2 = src[lane + 32];
    w1 = qw[lane];  w2 = qw[lane + 32];
    dst = Qo + (((size_t)b * H_ + h) * T_ + t) * HD_;
  } else {
    long id2 = id - NQ;
    int h = (int)(id2 % HKV_); long bt = id2 / HKV_;
    t = (int)(bt % T_); b = (int)(bt / T_);
    const float* src = kvraw + ((size_t)b * T_ + t) * NKV_ + h * HD_;
    v1 = src[lane]; v2 = src[lane + 32];
    w1 = kw[lane];  w2 = kw[lane + 32];
    dst = Ko + (((size_t)b * HKV_ + h) * T_ + t) * HD_;
    float f1 = src[256 + lane], f2 = src[256 + lane + 32];
    __bf16* vt = Vt + (((size_t)b * HKV_ + h) * HD_) * T_;
    vt[(size_t)lane * T_ + t]        = (__bf16)f1;
    vt[(size_t)(lane + 32) * T_ + t] = (__bf16)f2;
  }
  float ss = v1 * v1 + v2 * v2;
#pragma unroll
  for (int m = 16; m >= 1; m >>= 1) ss += __shfl_xor(ss, m, 32);
  const float rn = rsqrtf(ss * (1.f / HD_) + EPS_);
  const float a = v1 * rn * w1, c2 = v2 * rn * w2;
  const float fr = (float)t * powf(10000.f, -(float)lane / 32.f);
  const float cs = cosf(fr), sn = sinf(fr);
  dst[lane]      = (__bf16)(a * cs + c2 * sn);
  dst[lane + 32] = (__bf16)(-a * sn + c2 * cs);
}

// -------- Flash attention: one wave per 16-row query tile --------
__global__ __launch_bounds__(128)
void flash_attn_wmma(const __bf16* __restrict__ Q, const __bf16* __restrict__ Kc,
                     const __bf16* __restrict__ Vt, __bf16* __restrict__ Y) {
  __shared__ __bf16 sP[4][16][32];
  const int tid = threadIdx.x, wave = tid >> 5, lane = tid & 31;
  const int ln = lane & 15, hi = lane >> 4;
  const int wid = blockIdx.x * 4 + wave;
  const int qt = wid & 63;
  const int h  = (wid >> 6) & 15;
  const int b  = wid >> 10;
  const int t0 = qt * 16;
  const int hkv = h >> 2;

  const __bf16* qptr = Q  + (((size_t)b * H_ + h) * T_ + t0) * HD_;
  const __bf16* kptr = Kc + (((size_t)b * HKV_ + hkv) * T_) * HD_;
  const __bf16* vptr = Vt + (((size_t)b * HKV_ + hkv) * HD_) * T_;

  const v16bf qa0 = load_frag_a(qptr, HD_);
  const v16bf qa1 = load_frag_a(qptr + 32, HD_);

  const v8f vzero = {0.f,0.f,0.f,0.f,0.f,0.f,0.f,0.f};
  v8f o[4];
  float mrow[8], lrow[8];
#pragma unroll
  for (int j = 0; j < 4; ++j) o[j] = vzero;
#pragma unroll
  for (int r = 0; r < 8; ++r) { mrow[r] = -INFINITY; lrow[r] = 0.f; }

  const int nkb = (t0 + 47) >> 5;
  for (int kb = 0; kb < nkb; ++kb) {
    const int kbase = kb * 32;
    float pm[2][8];
#pragma unroll
    for (int nt = 0; nt < 2; ++nt) {
      v16bf b0 = load_frag_b(kptr + (size_t)(kbase + nt * 16) * HD_, HD_);
      v16bf b1 = load_frag_b(kptr + (size_t)(kbase + nt * 16) * HD_ + 32, HD_);
      v8f s = vzero;
      s = wmma_bf16(qa0, b0, s);
      s = wmma_bf16(qa1, b1, s);
      const int key = kbase + nt * 16 + ln;
#pragma unroll
      for (int r = 0; r < 8; ++r) {
        float v = s[r] * 0.125f;
        v = CAP_ * tanhf(v * (1.f / CAP_));
        const int qrow = t0 + hi * 8 + r;
        pm[nt][r] = (key > qrow) ? -INFINITY : v;
      }
    }
    float alpha[8];
#pragma unroll
    for (int r = 0; r < 8; ++r) {
      float mx = fmaxf(pm[0][r], pm[1][r]);
#pragma unroll
      for (int m = 8; m >= 1; m >>= 1) mx = fmaxf(mx, __shfl_xor(mx, m, 32));
      const float nm = fmaxf(mrow[r], mx);
      alpha[r] = expf(mrow[r] - nm);
      mrow[r] = nm;
      const float p0 = expf(pm[0][r] - nm);
      const float p1 = expf(pm[1][r] - nm);
      float rs = p0 + p1;
#pragma unroll
      for (int m = 8; m >= 1; m >>= 1) rs += __shfl_xor(rs, m, 32);
      lrow[r] = lrow[r] * alpha[r] + rs;
      sP[wave][hi * 8 + r][ln]      = (__bf16)p0;
      sP[wave][hi * 8 + r][ln + 16] = (__bf16)p1;
    }
    asm volatile("s_wait_dscnt 0" ::: "memory");
    const v16bf pa = load_frag_a(&sP[wave][0][0], 32);
#pragma unroll
    for (int j = 0; j < 4; ++j) {
#pragma unroll
      for (int r = 0; r < 8; ++r) o[j][r] = o[j][r] * alpha[r];
      v16bf vb = load_frag_b(vptr + (size_t)(j * 16) * T_ + kbase, T_);
      o[j] = wmma_bf16(pa, vb, o[j]);
    }
  }
  __bf16* yp = Y + ((size_t)b * T_ + t0) * C_ + h * HD_;
#pragma unroll
  for (int r = 0; r < 8; ++r) {
    const float inv = 1.f / lrow[r];
#pragma unroll
    for (int j = 0; j < 4; ++j)
      yp[(size_t)(hi * 8 + r) * C_ + j * 16 + ln] = (__bf16)(o[j][r] * inv);
  }
}

extern "C" void kernel_launch(void* const* d_in, const int* in_sizes, int n_in,
                              void* d_out, int out_size, void* d_ws, size_t ws_size,
                              hipStream_t stream) {
  (void)in_sizes; (void)n_in; (void)out_size; (void)ws_size;
  const float* x    = (const float*)d_in[0];
  const float* w_q  = (const float*)d_in[1];
  const float* w_kv = (const float*)d_in[2];
  const float* w_c  = (const float*)d_in[3];
  const float* qw   = (const float*)d_in[4];
  const float* kw   = (const float*)d_in[5];

  char* base = (char*)d_ws;
  size_t off = 0;
  auto take = [&](size_t bytes) -> char* {
    char* p = base + off;
    off += (bytes + 255) & ~(size_t)255;
    return p;
  };
  __bf16* xbf   = (__bf16*)take((size_t)B_ * T_ * C_ * 2);
  __bf16* wqb   = (__bf16*)take((size_t)C_ * C_ * 2);
  __bf16* wkvb  = (__bf16*)take((size_t)NKV_ * C_ * 2);
  __bf16* wcb   = (__bf16*)take((size_t)C_ * C_ * 2);
  float*  qraw  = (float*)take((size_t)B_ * T_ * C_ * 4);
  float*  kvraw = (float*)take((size_t)B_ * T_ * NKV_ * 4);
  __bf16* qbf   = (__bf16*)take((size_t)B_ * H_ * T_ * HD_ * 2);
  __bf16* kbf   = (__bf16*)take((size_t)B_ * HKV_ * T_ * HD_ * 2);
  __bf16* vtb   = (__bf16*)take((size_t)B_ * HKV_ * HD_ * T_ * 2);
  __bf16* ybf   = (__bf16*)take((size_t)B_ * T_ * C_ * 2);

  const int M = B_ * T_;
  // 1) converts (4 elems/thread)
  {
    int n4 = (B_ * T_ * C_) / 4;
    cvt_f32_bf16<<<(n4 + 255) / 256, 256, 0, stream>>>(x, xbf, n4);
    n4 = (C_ * C_) / 4;
    cvt_f32_bf16<<<(n4 + 255) / 256, 256, 0, stream>>>(w_q, wqb, n4);
    cvt_f32_bf16<<<(n4 + 255) / 256, 256, 0, stream>>>(w_c, wcb, n4);
    n4 = (NKV_ * C_) / 4;
    cvt_f32_bf16<<<(n4 + 255) / 256, 256, 0, stream>>>(w_kv, wkvb, n4);
  }
  // 2) projections (TDM-fed WMMA GEMMs, 128x128 tiles)
  gemm_bf16_wmma<<<dim3(C_ / 128, M / 128), 256, 0, stream>>>(xbf, wqb, qraw, M, C_, C_);
  gemm_bf16_wmma<<<dim3(NKV_ / 128, M / 128), 256, 0, stream>>>(xbf, wkvb, kvraw, M, NKV_, C_);
  // 3) fused RMSNorm + RoPE + pack (V transposed)
  norm_rope_pack<<<(B_ * T_ * (H_ + HKV_)) / 8, 256, 0, stream>>>(
      qraw, kvraw, qw, kw, qbf, kbf, vtb);
  // 4) flash attention (WMMA for QK^T and PV)
  flash_attn_wmma<<<(B_ * H_ * (T_ / 16)) / 4, 128, 0, stream>>>(qbf, kbf, vtb, ybf);
  // 5) output projection -> d_out (fp32)
  gemm_bf16_wmma<<<dim3(C_ / 128, M / 128), 256, 0, stream>>>(ybf, wcb, (float*)d_out, M, C_, C_);
}
